// GNNWithAgentPolicy_91268055040566
// MI455X (gfx1250) — matrix operand order repaired and checked by reading, not compiled, not run on hardware
//
#include <hip/hip_runtime.h>
#include <hip/hip_bf16.h>

#define NN_NODES  50000
#define NN_EDGES  800000
#define DIM       128
#define NN_AGENTS 1024
#define OUT_DIM   512   // HORIZON*ACT

typedef __attribute__((ext_vector_type(16))) _Float16 v16h;
typedef __attribute__((ext_vector_type(8)))  float    v8f;

// ---------------------------------------------------------------------------
// Edge scatter: aggr[dst] += x[src]  (128-dim rows). L2-resident f32 atomics.
// ---------------------------------------------------------------------------
#define EDGES_PER_BLOCK 32

__global__ void scatter_add_kernel(const float* __restrict__ x,
                                   const int* __restrict__ src,
                                   const int* __restrict__ dst,
                                   float* __restrict__ aggr,
                                   int n_edges) {
    const int t  = threadIdx.x;                 // feature dim 0..127
    const int e0 = blockIdx.x * EDGES_PER_BLOCK;
#pragma unroll 4
    for (int i = 0; i < EDGES_PER_BLOCK; ++i) {
        const int e = e0 + i;
        if (e < n_edges) {
            const int s = src[e];
            const int d = dst[e];
            const float v = x[(size_t)s * DIM + t];
            unsafeAtomicAdd(&aggr[(size_t)d * DIM + t], v);  // global_atomic_add_f32
        }
    }
}

// ---------------------------------------------------------------------------
// Generic WMMA GEMM: C = act( A1 @ W1^T [+ A2 @ W2^T] + bias )
//   A: [M,K] f32 (optionally row-gathered via rowidx), W: [N,K] f32 row-major
//   One wave per 16x16 output tile; K looped in steps of 32 via
//   v_wmma_f32_16x16x32_f16 (f32 accumulate).
// ---------------------------------------------------------------------------
__device__ __forceinline__ v16h load_a_frag(const float* __restrict__ A,
                                            size_t row, int K, int k0, int hi) {
    // 16-bit A 16x32 layout: lane<16 -> K {k0+0..7, k0+16..23};
    //                        lane>=16 -> K {k0+8..15, k0+24..31}
    const float* p = A + row * (size_t)K + k0 + hi * 8;
    const float4 q0 = ((const float4*)p)[0];
    const float4 q1 = ((const float4*)p)[1];
    const float4 q2 = ((const float4*)(p + 16))[0];
    const float4 q3 = ((const float4*)(p + 16))[1];
    v16h a;
    a[0]=(_Float16)q0.x; a[1]=(_Float16)q0.y; a[2]=(_Float16)q0.z; a[3]=(_Float16)q0.w;
    a[4]=(_Float16)q1.x; a[5]=(_Float16)q1.y; a[6]=(_Float16)q1.z; a[7]=(_Float16)q1.w;
    a[8]=(_Float16)q2.x; a[9]=(_Float16)q2.y; a[10]=(_Float16)q2.z; a[11]=(_Float16)q2.w;
    a[12]=(_Float16)q3.x; a[13]=(_Float16)q3.y; a[14]=(_Float16)q3.z; a[15]=(_Float16)q3.w;
    return a;
}

__device__ __forceinline__ v16h load_b_frag(const float* __restrict__ W,
                                            int n, int K, int k0, int hi) {
    // 16-bit B 32x16 layout: lane n(<16) holds K {k0+0..7, k0+16..23} of column n
    //                        lane n+16 holds K {k0+8..15, k0+24..31} of column n.
    // B column n == row n of W ([N,K] row-major).
    const float* p = W + (size_t)n * K + k0 + hi * 8;
    const float4 q0 = ((const float4*)p)[0];
    const float4 q1 = ((const float4*)p)[1];
    const float4 q2 = ((const float4*)(p + 16))[0];
    const float4 q3 = ((const float4*)(p + 16))[1];
    v16h b;
    b[0]=(_Float16)q0.x; b[1]=(_Float16)q0.y; b[2]=(_Float16)q0.z; b[3]=(_Float16)q0.w;
    b[4]=(_Float16)q1.x; b[5]=(_Float16)q1.y; b[6]=(_Float16)q1.z; b[7]=(_Float16)q1.w;
    b[8]=(_Float16)q2.x; b[9]=(_Float16)q2.y; b[10]=(_Float16)q2.z; b[11]=(_Float16)q2.w;
    b[12]=(_Float16)q3.x; b[13]=(_Float16)q3.y; b[14]=(_Float16)q3.z; b[15]=(_Float16)q3.w;
    return b;
}

__global__ void wmma_gemm_kernel(const float* __restrict__ A,
                                 const float* __restrict__ W,
                                 const float* __restrict__ bias,
                                 const float* __restrict__ A2,
                                 const float* __restrict__ W2,
                                 const int* __restrict__ rowidx,
                                 float* __restrict__ C,
                                 int M, int N, int K, int relu) {
    const int lane    = threadIdx.x & 31;
    const int wave    = threadIdx.x >> 5;
    const int tiles_n = N >> 4;
    const int tile    = blockIdx.x * 8 + wave;      // 8 waves per block
    const int total   = (M >> 4) * tiles_n;
    if (tile >= total) return;                       // wave-uniform; EXEC stays full

    const int mtile = tile / tiles_n;
    const int ntile = tile - mtile * tiles_n;
    const int lh    = lane & 15;
    const int hi    = lane >> 4;

    const int    arow_g = mtile * 16 + lh;           // A row for this lane
    const size_t arow   = rowidx ? (size_t)rowidx[arow_g] : (size_t)arow_g;
    const int    ncol   = ntile * 16 + lh;           // output col / W row

    v8f acc = {};
    for (int k0 = 0; k0 < K; k0 += 32) {
        v16h a = load_a_frag(A, arow, K, k0, hi);
        v16h b = load_b_frag(W, ncol, K, k0, hi);
        acc = __builtin_amdgcn_wmma_f32_16x16x32_f16(false, a, false, b,
                                                     (short)0, acc, false, false);
    }
    if (A2) {  // fused second term (GraphConv root path), same K
        for (int k0 = 0; k0 < K; k0 += 32) {
            v16h a = load_a_frag(A2, arow, K, k0, hi);
            v16h b = load_b_frag(W2, ncol, K, k0, hi);
            acc = __builtin_amdgcn_wmma_f32_16x16x32_f16(false, a, false, b,
                                                         (short)0, acc, false, false);
        }
    }

    const float bv = bias ? bias[ncol] : 0.0f;
#pragma unroll
    for (int r = 0; r < 8; ++r) {
        float v = acc[r] + bv;
        if (relu) v = fmaxf(v, 0.0f);
        const int mrow = mtile * 16 + hi * 8 + r;    // C layout: vgpr r -> M=r / M=8+r
        C[(size_t)mrow * N + ncol] = v;
    }
}

static inline void run_gemm(hipStream_t s, const float* A, const float* W,
                            const float* bias, const float* A2, const float* W2,
                            const int* rowidx, float* C, int M, int N, int K, int relu) {
    const int tiles  = (M / 16) * (N / 16);
    const int blocks = (tiles + 7) / 8;
    wmma_gemm_kernel<<<blocks, 256, 0, s>>>(A, W, bias, A2, W2, rowidx, C, M, N, K, relu);
}

// ---------------------------------------------------------------------------
extern "C" void kernel_launch(void* const* d_in, const int* in_sizes, int n_in,
                              void* d_out, int out_size, void* d_ws, size_t ws_size,
                              hipStream_t stream) {
    const float* x      = (const float*)d_in[0];
    const int*   eidx   = (const int*)d_in[1];
    const int*   src    = eidx;
    const int*   dst    = eidx + NN_EDGES;
    const int*   agent  = (const int*)d_in[2];
    const float* W1_rel = (const float*)d_in[3];
    const float* b1_rel = (const float*)d_in[4];
    const float* W1_root= (const float*)d_in[5];
    const float* W2_rel = (const float*)d_in[6];
    const float* b2_rel = (const float*)d_in[7];
    const float* W2_root= (const float*)d_in[8];
    const float* Wp1    = (const float*)d_in[9];
    const float* bp1    = (const float*)d_in[10];
    const float* Wp2    = (const float*)d_in[11];
    const float* bp2    = (const float*)d_in[12];
    const float* Wp3    = (const float*)d_in[13];
    const float* bp3    = (const float*)d_in[14];
    float*       out    = (float*)d_out;

    // workspace partition (floats): aggr | h | z | m1 | m2
    float* aggr = (float*)d_ws;
    float* h    = aggr + (size_t)NN_NODES * DIM;
    float* z    = h    + (size_t)NN_NODES * DIM;
    float* m1   = z    + (size_t)NN_NODES * DIM;
    float* m2   = m1   + (size_t)NN_AGENTS * DIM;

    const size_t aggrBytes   = (size_t)NN_NODES * DIM * sizeof(float);
    const int    scatterGrid = (NN_EDGES + EDGES_PER_BLOCK - 1) / EDGES_PER_BLOCK;

    // ---- GraphConv 1: h = relu(aggr(x) @ W1_rel^T + b1 + x @ W1_root^T) ----
    hipMemsetAsync(aggr, 0, aggrBytes, stream);
    scatter_add_kernel<<<scatterGrid, DIM, 0, stream>>>(x, src, dst, aggr, NN_EDGES);
    run_gemm(stream, aggr, W1_rel, b1_rel, x, W1_root, nullptr,
             h, NN_NODES, DIM, DIM, 1);

    // ---- GraphConv 2: z = relu(aggr(h) @ W2_rel^T + b2 + h @ W2_root^T) ----
    hipMemsetAsync(aggr, 0, aggrBytes, stream);
    scatter_add_kernel<<<scatterGrid, DIM, 0, stream>>>(h, src, dst, aggr, NN_EDGES);
    run_gemm(stream, aggr, W2_rel, b2_rel, h, W2_root, nullptr,
             z, NN_NODES, DIM, DIM, 1);

    // ---- Policy MLP on gathered agents ----
    run_gemm(stream, z,  Wp1, bp1, nullptr, nullptr, agent,   m1, NN_AGENTS, DIM,     DIM, 1);
    run_gemm(stream, m1, Wp2, bp2, nullptr, nullptr, nullptr, m2, NN_AGENTS, DIM,     DIM, 1);
    run_gemm(stream, m2, Wp3, bp3, nullptr, nullptr, nullptr, out, NN_AGENTS, OUT_DIM, DIM, 0);
}
